// QuerySegHead_44770739093795
// MI455X (gfx1250) — matrix operand order repaired and checked by reading, not compiled, not run on hardware
//
#include <hip/hip_runtime.h>
#include <hip/hip_bf16.h>
#include <math.h>

// ---------------- problem constants ----------------
#define B_    2
#define C_    512
#define NS_   4
#define NQ_   256
#define NF_   1024
#define NP_   24576
#define NCLS_ 13
#define CS_   128

// ---------------- WMMA / vector types (gfx1250, wave32) ----------------
typedef __attribute__((ext_vector_type(16))) __bf16 v16bf;
typedef __attribute__((ext_vector_type(8)))  __bf16 v8bf;
typedef __attribute__((ext_vector_type(4)))  __bf16 v4bf;
typedef __attribute__((ext_vector_type(8)))  float  v8f;
typedef __attribute__((ext_vector_type(4)))  float  v4f;

__device__ __forceinline__ float gelu_tanh(float x) {
    const float c0 = 0.7978845608028654f;   // sqrt(2/pi)
    const float c1 = 0.044715f;
    float x3 = x * x * x;
    return 0.5f * x * (1.0f + tanhf(c0 * (x + c1 * x3)));
}

// ============================================================
// Generic batched WMMA GEMM:  D[b,m,n] = epilogue( alpha * sum_k A[b,m,k]*B[b,k,n] )
// Tile 64(M) x 128(N) x 32(K); 256 threads = 8 waves; 4 WMMAs per wave per K-step.
// LDS: As[m][k] (k-contig), Bst[n][k] (k-contig, transposed) -> ds_load_b128 fragments.
// Vectorized global load fast paths for all full tiles in this workload.
// ============================================================
struct GemmArgs {
    const void* A; long long a_bs, a_ms, a_ks; int a_bf16;
    const void* Bp; long long b_bs, b_ks, b_ns; int b_bf16;
    void* D; long long d_bs, d_ms, d_ns; int d_bf16;
    const float* resid; long long r_bs, r_ms, r_ns;
    const float* bias; const float* bn_g; const float* bn_be;
    int vec_axis;    // 0 = epilogue vectors indexed by m, 1 = by n
    int M, N, K;
    float alpha;
    int do_gelu;
};

#define BM 64
#define BN 128
#define BK 32

__global__ __launch_bounds__(256) void wmma_gemm_kernel(GemmArgs g) {
    __shared__ __bf16 As[BM * BK];    // [m][k], row stride 32
    __shared__ __bf16 Bst[BN * BK];   // [n][k], row stride 32 (transposed tile)

    const int b   = blockIdx.z;
    const int m0  = blockIdx.y * BM;
    const int n0  = blockIdx.x * BN;
    const int tid = threadIdx.x;
    const int ln  = tid & 31;
    const int wv  = tid >> 5;
    const int li  = ln & 15;      // lane index within half
    const int hi  = ln >> 4;      // wave half

    const int wm = (wv & 3) * 16;     // wave m sub-tile
    const int wnb = (wv >> 2) * 64;   // wave n range: wnb .. wnb+63 (4 tiles of 16)

    const bool full = (m0 + BM <= g.M) && (n0 + BN <= g.N) && (g.K % BK == 0);
    // uniform fast-path selectors
    const int pathA = !full ? 0 : (g.a_ks == 1 ? (g.a_bf16 ? 2 : 1)
                          : ((g.a_ms == 1 && !g.a_bf16) ? 3 : 0));
    const int pathB = !full ? 0 : ((g.b_ks == 1 && !g.b_bf16) ? 1
                          : (g.b_ns == 1 ? (g.b_bf16 ? 3 : 2) : 0));

    v8f acc[4] = {};

    for (int k0 = 0; k0 < g.K; k0 += BK) {
        // ================= A tile: BM x BK =================
        if (pathA == 1) {           // f32, k-contiguous: float4 loads, v4bf stores
            #pragma unroll
            for (int t = 0; t < 2; ++t) {
                int e = tid + t * 256;                 // 512 vec4 total
                int am = e >> 3, ak = (e & 7) << 2;
                const float* src = (const float*)g.A + (long long)b * g.a_bs
                                 + (long long)(m0 + am) * g.a_ms + (k0 + ak);
                v4f v = *(const v4f*)src;
                v4bf o; o[0] = (__bf16)v[0]; o[1] = (__bf16)v[1];
                o[2] = (__bf16)v[2]; o[3] = (__bf16)v[3];
                *(v4bf*)(As + am * BK + ak) = o;
            }
        } else if (pathA == 2) {    // bf16, k-contiguous: v8bf load/store, 1 per thread
            int am = tid >> 2, ak = (tid & 3) << 3;
            const __bf16* src = (const __bf16*)g.A + (long long)b * g.a_bs
                              + (long long)(m0 + am) * g.a_ms + (k0 + ak);
            *(v8bf*)(As + am * BK + ak) = *(const v8bf*)src;
        } else if (pathA == 3) {    // f32, m-contiguous: float4 along m, scatter b16
            #pragma unroll
            for (int t = 0; t < 2; ++t) {
                int e = tid + t * 256;
                int ak = e >> 4, am = (e & 15) << 2;   // 16 vec4 per k-column
                const float* src = (const float*)g.A + (long long)b * g.a_bs
                                 + (m0 + am) + (long long)(k0 + ak) * g.a_ks;
                v4f v = *(const v4f*)src;
                #pragma unroll
                for (int j = 0; j < 4; ++j) As[(am + j) * BK + ak] = (__bf16)v[j];
            }
        } else {                    // generic, bounds-checked
            #pragma unroll
            for (int t = 0; t < (BM * BK) / 256; ++t) {
                int e = tid + t * 256;
                int am = e / BK, ak = e % BK;
                int gm = m0 + am, gk = k0 + ak;
                float v = 0.0f;
                if (gm < g.M && gk < g.K) {
                    long long off = (long long)b * g.a_bs + (long long)gm * g.a_ms
                                  + (long long)gk * g.a_ks;
                    v = g.a_bf16 ? (float)((const __bf16*)g.A)[off]
                                 : ((const float*)g.A)[off];
                }
                As[am * BK + ak] = (__bf16)v;
            }
        }

        // ================= B tile: BK x BN -> Bst[n][k] =================
        if (pathB == 1) {           // f32, k-contiguous (weights [n,K]): float4 -> v4bf
            #pragma unroll
            for (int t = 0; t < 4; ++t) {
                int e = tid + t * 256;                 // 1024 vec4 total
                int bn = e >> 3, bk = (e & 7) << 2;
                const float* src = (const float*)g.Bp + (long long)b * g.b_bs
                                 + (k0 + bk) + (long long)(n0 + bn) * g.b_ns;
                v4f v = *(const v4f*)src;
                v4bf o; o[0] = (__bf16)v[0]; o[1] = (__bf16)v[1];
                o[2] = (__bf16)v[2]; o[3] = (__bf16)v[3];
                *(v4bf*)(Bst + bn * BK + bk) = o;
            }
        } else if (pathB == 2) {    // f32, n-contiguous: float4 along n, scatter b16
            #pragma unroll
            for (int t = 0; t < 4; ++t) {
                int e = tid + t * 256;
                int bk = e >> 5, bn = (e & 31) << 2;   // 32 vec4 per k-row
                const float* src = (const float*)g.Bp + (long long)b * g.b_bs
                                 + (long long)(k0 + bk) * g.b_ks + (n0 + bn);
                v4f v = *(const v4f*)src;
                #pragma unroll
                for (int j = 0; j < 4; ++j) Bst[(bn + j) * BK + bk] = (__bf16)v[j];
            }
        } else if (pathB == 3) {    // bf16, n-contiguous: v8bf along n, scatter b16
            #pragma unroll
            for (int t = 0; t < 2; ++t) {
                int e = tid + t * 256;                 // 512 vec8 total
                int bk = e >> 4, bn = (e & 15) << 3;
                const __bf16* src = (const __bf16*)g.Bp + (long long)b * g.b_bs
                                  + (long long)(k0 + bk) * g.b_ks + (n0 + bn);
                v8bf v = *(const v8bf*)src;
                #pragma unroll
                for (int j = 0; j < 8; ++j) Bst[(bn + j) * BK + bk] = v[j];
            }
        } else {                    // generic, bounds-checked
            #pragma unroll
            for (int t = 0; t < (BK * BN) / 256; ++t) {
                int e = tid + t * 256;
                int bk = e / BN, bn = e % BN;
                int gk = k0 + bk, gn = n0 + bn;
                float v = 0.0f;
                if (gk < g.K && gn < g.N) {
                    long long off = (long long)b * g.b_bs + (long long)gk * g.b_ks
                                  + (long long)gn * g.b_ns;
                    v = g.b_bf16 ? (float)((const __bf16*)g.Bp)[off]
                                 : ((const float*)g.Bp)[off];
                }
                Bst[bn * BK + bk] = (__bf16)v;
            }
        }
        __syncthreads();

        // ---- fragments via ds_load_b128 ----
        // A 16x32: lanes 0-15 rows 0-15 K 0..15; lanes 16-31 same rows K 16..31
        const v8bf* ap = (const v8bf*)(As + (wm + li) * BK);
        v8bf a_lo = ap[2 * hi];
        v8bf a_hi = ap[2 * hi + 1];
        v16bf af = __builtin_shufflevector(a_lo, a_hi,
                    0, 1, 2, 3, 4, 5, 6, 7, 8, 9, 10, 11, 12, 13, 14, 15);

        #pragma unroll
        for (int j = 0; j < 4; ++j) {
            // B 32x16: lanes 0-15 cols, K 0..15; lanes 16-31 same cols, K 16..31
            const v8bf* bp = (const v8bf*)(Bst + (wnb + j * 16 + li) * BK);
            v8bf b_lo = bp[2 * hi];
            v8bf b_hi = bp[2 * hi + 1];
            v16bf bfj = __builtin_shufflevector(b_lo, b_hi,
                        0, 1, 2, 3, 4, 5, 6, 7, 8, 9, 10, 11, 12, 13, 14, 15);
            acc[j] = __builtin_amdgcn_wmma_f32_16x16x32_bf16(
                         false, af, false, bfj, (short)0, acc[j], false, false);
        }
        __syncthreads();
    }

    // ---- epilogue; D 16x16 layout: lane -> n = li, vgpr v -> m = v + 8*hi ----
    #pragma unroll
    for (int j = 0; j < 4; ++j) {
        int gn = n0 + wnb + j * 16 + li;
        #pragma unroll
        for (int v = 0; v < 8; ++v) {
            int gm = m0 + wm + v + 8 * hi;
            if (gm >= g.M || gn >= g.N) continue;
            float x = acc[j][v] * g.alpha;
            int vi = g.vec_axis ? gn : gm;
            if (g.bias) x += g.bias[vi];
            if (g.bn_g) x = x * g.bn_g[vi] + g.bn_be[vi];
            if (g.do_gelu) x = gelu_tanh(x);
            if (g.resid)
                x += g.resid[(long long)b * g.r_bs + (long long)gm * g.r_ms
                             + (long long)gn * g.r_ns];
            long long off = (long long)b * g.d_bs + (long long)gm * g.d_ms
                          + (long long)gn * g.d_ns;
            if (g.d_bf16) ((__bf16*)g.D)[off] = (__bf16)x;
            else          ((float*)g.D)[off]  = x;
        }
    }
}

// ============================================================
// argmin over NP original points per query  (block = one query)
// ============================================================
__global__ __launch_bounds__(256) void argmin_query_kernel(const float* __restrict__ op,
                                                           const float* __restrict__ query,
                                                           float* __restrict__ out_qidx) {
    __shared__ float sd[256];
    __shared__ int   si[256];
    const int b = blockIdx.y, q = blockIdx.x, tid = threadIdx.x;
    const float qx = query[((long long)b * 3 + 0) * NQ_ + q];
    const float qy = query[((long long)b * 3 + 1) * NQ_ + q];
    const float qz = query[((long long)b * 3 + 2) * NQ_ + q];
    const float* px = op + (long long)b * 3 * NP_;
    float bd = 3.4e38f; int bi = 0;
    for (int p = tid; p < NP_; p += 256) {
        float dx = px[p] - qx, dy = px[NP_ + p] - qy, dz = px[2 * NP_ + p] - qz;
        float d = dx * dx + dy * dy + dz * dz;
        if (d < bd || (d == bd && p < bi)) { bd = d; bi = p; }
    }
    sd[tid] = bd; si[tid] = bi; __syncthreads();
    for (int s = 128; s > 0; s >>= 1) {
        if (tid < s) {
            if (sd[tid + s] < sd[tid] || (sd[tid + s] == sd[tid] && si[tid + s] < si[tid])) {
                sd[tid] = sd[tid + s]; si[tid] = si[tid + s];
            }
        }
        __syncthreads();
    }
    if (tid == 0) out_qidx[(long long)b * NQ_ + q] = (float)si[0];
}

// ============================================================
// three_nn: top-3 nearest downsampled points + inverse-distance weights
// ============================================================
__global__ __launch_bounds__(256) void three_nn_kernel(const float* __restrict__ op,
                                                       const float* __restrict__ pts,
                                                       int* __restrict__ idx_out,
                                                       float* __restrict__ w_out) {
    __shared__ float px[NF_], py[NF_], pz[NF_];
    const int b = blockIdx.y, tid = threadIdx.x;
    for (int i = tid; i < NF_; i += 256) {
        px[i] = pts[((long long)b * 3 + 0) * NF_ + i];
        py[i] = pts[((long long)b * 3 + 1) * NF_ + i];
        pz[i] = pts[((long long)b * 3 + 2) * NF_ + i];
    }
    __syncthreads();
    const int p = blockIdx.x * 256 + tid;
    if (p >= NP_) return;
    const float ox = op[((long long)b * 3 + 0) * NP_ + p];
    const float oy = op[((long long)b * 3 + 1) * NP_ + p];
    const float oz = op[((long long)b * 3 + 2) * NP_ + p];
    float d0 = 3.4e38f, d1 = 3.4e38f, d2 = 3.4e38f;
    int i0 = 0, i1 = 0, i2 = 0;
    for (int j = 0; j < NF_; ++j) {
        float dx = ox - px[j], dy = oy - py[j], dz = oz - pz[j];
        float d = dx * dx + dy * dy + dz * dz;
        if (d < d0)      { d2 = d1; i2 = i1; d1 = d0; i1 = i0; d0 = d; i0 = j; }
        else if (d < d1) { d2 = d1; i2 = i1; d1 = d; i1 = j; }
        else if (d < d2) { d2 = d; i2 = j; }
    }
    float r0 = 1.0f / (fmaxf(d0, 0.0f) + 1e-8f);
    float r1 = 1.0f / (fmaxf(d1, 0.0f) + 1e-8f);
    float r2 = 1.0f / (fmaxf(d2, 0.0f) + 1e-8f);
    float rs = r0 + r1 + r2;
    long long o = ((long long)b * NP_ + p) * 3;
    idx_out[o] = i0; idx_out[o + 1] = i1; idx_out[o + 2] = i2;
    w_out[o] = r0 / rs; w_out[o + 1] = r1 / rs; w_out[o + 2] = r2 / rs;
}

// ============================================================
// row softmax (row length == blockDim.x == NQ_)
// ============================================================
__global__ __launch_bounds__(256) void softmax_rows_kernel(float* __restrict__ S) {
    __shared__ float red[256];
    const int tid = threadIdx.x;
    float* row = S + (long long)blockIdx.x * NQ_;
    float x = row[tid];
    red[tid] = x; __syncthreads();
    for (int s = 128; s > 0; s >>= 1) { if (tid < s) red[tid] = fmaxf(red[tid], red[tid + s]); __syncthreads(); }
    float mx = red[0]; __syncthreads();
    float e = __expf(x - mx);
    red[tid] = e; __syncthreads();
    for (int s = 128; s > 0; s >>= 1) { if (tid < s) red[tid] += red[tid + s]; __syncthreads(); }
    row[tid] = e / red[0];
}

// ============================================================
// interp base gather: interp[b,o,p] = sum_k w[b,p,k] * feats[b,o,idx[b,p,k]]
// ============================================================
__global__ __launch_bounds__(256) void gather_base_kernel(const float* __restrict__ feats,
                                                          const int* __restrict__ idx,
                                                          const float* __restrict__ w,
                                                          float* __restrict__ out) {
    long long t = (long long)blockIdx.x * 256 + threadIdx.x;
    if (t >= (long long)B_ * C_ * NP_) return;
    int p = (int)(t % NP_);
    long long r = t / NP_;
    int o = (int)(r % C_);
    int b = (int)(r / C_);
    long long iw = ((long long)b * NP_ + p) * 3;
    int i0 = idx[iw], i1 = idx[iw + 1], i2 = idx[iw + 2];
    float w0 = w[iw], w1 = w[iw + 1], w2 = w[iw + 2];
    const float* f = feats + ((long long)b * C_ + o) * NF_;
    out[t] = w0 * f[i0] + w1 * f[i1] + w2 * f[i2];
}

// ============================================================
// stage h1: gather from U = W1@pf (interpolation is linear), BN+GELU, bf16 out
// ============================================================
__global__ __launch_bounds__(256) void gather_h1_kernel(const float* __restrict__ U,
                                                        const int* __restrict__ idx,
                                                        const float* __restrict__ w,
                                                        const float* __restrict__ b1,
                                                        const float* __restrict__ g1,
                                                        const float* __restrict__ be1,
                                                        __bf16* __restrict__ h1) {
    long long t = (long long)blockIdx.x * 256 + threadIdx.x;
    if (t >= (long long)B_ * C_ * NP_) return;
    int p = (int)(t % NP_);
    long long r = t / NP_;
    int o = (int)(r % C_);
    int b = (int)(r / C_);
    long long iw = ((long long)b * NP_ + p) * 3;
    int i0 = idx[iw], i1 = idx[iw + 1], i2 = idx[iw + 2];
    float w0 = w[iw], w1 = w[iw + 1], w2 = w[iw + 2];
    const float* f = U + ((long long)b * C_ + o) * NF_;
    float acc = w0 * f[i0] + w1 * f[i1] + w2 * f[i2];
    float x = gelu_tanh((acc + b1[o]) * g1[o] + be1[o]);
    h1[t] = (__bf16)x;
}

// ============================================================
// log_softmax over NCLS_ classes (thread per row)
// ============================================================
__global__ __launch_bounds__(256) void logsoftmax_kernel(const float* __restrict__ preds,
                                                         float* __restrict__ out, long long nrows) {
    long long r = (long long)blockIdx.x * 256 + threadIdx.x;
    if (r >= nrows) return;
    const float* x = preds + r * NCLS_;
    float mx = -3.4e38f;
    #pragma unroll
    for (int c = 0; c < NCLS_; ++c) mx = fmaxf(mx, x[c]);
    float s = 0.0f;
    #pragma unroll
    for (int c = 0; c < NCLS_; ++c) s += __expf(x[c] - mx);
    float lse = mx + logf(s);
    #pragma unroll
    for (int c = 0; c < NCLS_; ++c) out[r * NCLS_ + c] = x[c] - lse;
}

// ============================================================
// host: launch sequence
// ============================================================
static void launch_gemm(hipStream_t s, const GemmArgs& g, int nb) {
    dim3 grid((g.N + BN - 1) / BN, (g.M + BM - 1) / BM, nb);
    wmma_gemm_kernel<<<grid, 256, 0, s>>>(g);
}

extern "C" void kernel_launch(void* const* d_in, const int* in_sizes, int n_in,
                              void* d_out, int out_size, void* d_ws, size_t ws_size,
                              hipStream_t stream) {
    (void)in_sizes; (void)n_in; (void)out_size; (void)ws_size;

    // -------- inputs (setup_inputs order) --------
    const float* original_points = (const float*)d_in[0];   // [B,3,NP]
    const float* points          = (const float*)d_in[1];   // [B,3,NF]
    /* points_index (d_in[2]) unused by reference */
    const float* query           = (const float*)d_in[3];   // [B,3,NQ]
    const float* query_feats     = (const float*)d_in[4];   // [B,C,NQ]
    const float* points_feats    = (const float*)d_in[5];   // [B,C,NF]
    const float* pf[NS_] = { (const float*)d_in[6], (const float*)d_in[7],
                             (const float*)d_in[8], (const float*)d_in[9] };

    // -------- params: JAX tree-flatten of nested dict => sorted keys, starting at index 10 --------
    auto P = [&](int i) { return (const float*)d_in[i]; };
    const float* cls_W1 = P(10); const float* cls_W2 = P(11);
    const float* cls_b1 = P(12); const float* cls_b2 = P(13);
    const float* cls_be1 = P(14); const float* cls_g1 = P(15);
    // mlp{i}: base 16+8i : W1,W2,b1,b2,be1,be2,g1,g2
    const float* pcls_W1 = P(48); const float* pcls_W2 = P(49);
    const float* pcls_b1 = P(50); const float* pcls_b2 = P(51);
    const float* pcls_be1 = P(52); const float* pcls_g1 = P(53);
    const float* qcls_W1 = P(54); const float* qcls_W2 = P(55);
    const float* qcls_b1 = P(56); const float* qcls_b2 = P(57);
    const float* qcls_be1 = P(58); const float* qcls_g1 = P(59);
    const float* qfp_W = P(60); const float* qfp_b = P(61);

    // -------- output layout (flat concat, float) --------
    float* out = (float*)d_out;
    float* out_qpreds = out;                                      // 2*256*13
    float* out_qidx   = out + (long long)B_ * NQ_ * NCLS_;        // 2*256
    float* out_ppreds = out_qidx + (long long)B_ * NQ_;           // 2*1024*13
    float* out_preds  = out_ppreds + (long long)B_ * NF_ * NCLS_; // 2*24576*13
    float* out_logsm  = out_preds + (long long)B_ * NP_ * NCLS_;  // 2*24576*13

    // -------- workspace carve-up --------
    char* ws = (char*)d_ws;
    size_t off = 0;
    auto carve = [&](size_t bytes) { char* p = ws + off; off = (off + bytes + 255) & ~(size_t)255; return p; };
    float*  S      = (float*) carve((size_t)B_ * NF_ * NQ_ * 4);
    float*  Hq     = (float*) carve((size_t)B_ * NQ_ * 256 * 4);
    float*  fused  = (float*) carve((size_t)B_ * C_ * NF_ * 4);
    float*  pfnew  = (float*) carve((size_t)B_ * C_ * NF_ * 4);
    float*  Hp     = (float*) carve((size_t)B_ * NF_ * 256 * 4);
    int*    nnidx  = (int*)   carve((size_t)B_ * NP_ * 3 * 4);
    float*  nnw    = (float*) carve((size_t)B_ * NP_ * 3 * 4);
    float*  U      = (float*) carve((size_t)B_ * C_ * NF_ * 4);
    float*  interp = (float*) carve((size_t)B_ * C_ * NP_ * 4);
    __bf16* h1     = (__bf16*)carve((size_t)B_ * C_ * NP_ * 2);  // reused later as Hc (bf16)
    __bf16* Hc     = h1;

    GemmArgs g;

    // ---- 1) query_index ----
    argmin_query_kernel<<<dim3(NQ_, B_), 256, 0, stream>>>(original_points, query, out_qidx);

    // ---- 2) query classifier ----
    g = GemmArgs{};
    g.A = query_feats; g.a_bs = (long long)C_ * NQ_; g.a_ms = 1; g.a_ks = NQ_;
    g.Bp = qcls_W1; g.b_bs = 0; g.b_ks = 1; g.b_ns = C_;
    g.D = Hq; g.d_bs = (long long)NQ_ * 256; g.d_ms = 256; g.d_ns = 1;
    g.bias = qcls_b1; g.bn_g = qcls_g1; g.bn_be = qcls_be1; g.vec_axis = 1;
    g.M = NQ_; g.N = 256; g.K = C_; g.alpha = 1.0f; g.do_gelu = 1;
    launch_gemm(stream, g, B_);

    g = GemmArgs{};
    g.A = Hq; g.a_bs = (long long)NQ_ * 256; g.a_ms = 256; g.a_ks = 1;
    g.Bp = qcls_W2; g.b_bs = 0; g.b_ks = 1; g.b_ns = 256;
    g.D = out_qpreds; g.d_bs = (long long)NQ_ * NCLS_; g.d_ms = NCLS_; g.d_ns = 1;
    g.bias = qcls_b2; g.vec_axis = 1;
    g.M = NQ_; g.N = NCLS_; g.K = 256; g.alpha = 1.0f;
    launch_gemm(stream, g, B_);

    // ---- 3) QFP attention: S = P^T Q / sqrt(C) ----
    g = GemmArgs{};
    g.A = points_feats; g.a_bs = (long long)C_ * NF_; g.a_ms = 1; g.a_ks = NF_;
    g.Bp = query_feats; g.b_bs = (long long)C_ * NQ_; g.b_ks = NQ_; g.b_ns = 1;
    g.D = S; g.d_bs = (long long)NF_ * NQ_; g.d_ms = NQ_; g.d_ns = 1;
    g.M = NF_; g.N = NQ_; g.K = C_; g.alpha = 0.04419417382415922f; // 1/sqrt(512)
    launch_gemm(stream, g, B_);

    softmax_rows_kernel<<<B_ * NF_, NQ_, 0, stream>>>(S);

    // fused[b,c,m] = sum_q qf[b,c,q] * attn[b,m,q]
    g = GemmArgs{};
    g.A = query_feats; g.a_bs = (long long)C_ * NQ_; g.a_ms = NQ_; g.a_ks = 1;
    g.Bp = S; g.b_bs = (long long)NF_ * NQ_; g.b_ks = 1; g.b_ns = NQ_;
    g.D = fused; g.d_bs = (long long)C_ * NF_; g.d_ms = NF_; g.d_ns = 1;
    g.M = C_; g.N = NF_; g.K = NQ_; g.alpha = 1.0f;
    launch_gemm(stream, g, B_);

    // pfnew = points_feats + gelu(qfp_W @ fused + qfp_b)
    g = GemmArgs{};
    g.A = qfp_W; g.a_bs = 0; g.a_ms = C_; g.a_ks = 1;
    g.Bp = fused; g.b_bs = (long long)C_ * NF_; g.b_ks = NF_; g.b_ns = 1;
    g.D = pfnew; g.d_bs = (long long)C_ * NF_; g.d_ms = NF_; g.d_ns = 1;
    g.bias = qfp_b; g.vec_axis = 0; g.do_gelu = 1;
    g.resid = points_feats; g.r_bs = (long long)C_ * NF_; g.r_ms = NF_; g.r_ns = 1;
    g.M = C_; g.N = NF_; g.K = C_; g.alpha = 1.0f;
    launch_gemm(stream, g, B_);

    // ---- 4) points classifier (on updated feats) ----
    g = GemmArgs{};
    g.A = pfnew; g.a_bs = (long long)C_ * NF_; g.a_ms = 1; g.a_ks = NF_;
    g.Bp = pcls_W1; g.b_bs = 0; g.b_ks = 1; g.b_ns = C_;
    g.D = Hp; g.d_bs = (long long)NF_ * 256; g.d_ms = 256; g.d_ns = 1;
    g.bias = pcls_b1; g.bn_g = pcls_g1; g.bn_be = pcls_be1; g.vec_axis = 1; g.do_gelu = 1;
    g.M = NF_; g.N = 256; g.K = C_; g.alpha = 1.0f;
    launch_gemm(stream, g, B_);

    g = GemmArgs{};
    g.A = Hp; g.a_bs = (long long)NF_ * 256; g.a_ms = 256; g.a_ks = 1;
    g.Bp = pcls_W2; g.b_bs = 0; g.b_ks = 1; g.b_ns = 256;
    g.D = out_ppreds; g.d_bs = (long long)NF_ * NCLS_; g.d_ms = NCLS_; g.d_ns = 1;
    g.bias = pcls_b2; g.vec_axis = 1;
    g.M = NF_; g.N = NCLS_; g.K = 256; g.alpha = 1.0f;
    launch_gemm(stream, g, B_);

    // ---- 5) three_nn + base interpolation ----
    three_nn_kernel<<<dim3((NP_ + 255) / 256, B_), 256, 0, stream>>>(original_points, points, nnidx, nnw);

    long long tot = (long long)B_ * C_ * NP_;
    int gblocks = (int)((tot + 255) / 256);
    gather_base_kernel<<<gblocks, 256, 0, stream>>>(pfnew, nnidx, nnw, interp);

    // ---- 6) per-stage MLPs: U = W1@pf; h1 = gelu(BN(interp3(U))); interp += gelu(BN(W2@h1 + b2)) ----
    for (int i = NS_ - 1; i >= 0; --i) {
        int base = 16 + 8 * i; // mlp{i}: W1,W2,b1,b2,be1,be2,g1,g2 (sorted)
        const float* W1 = P(base + 0); const float* W2 = P(base + 1);
        const float* b1 = P(base + 2); const float* b2 = P(base + 3);
        const float* be1 = P(base + 4); const float* be2 = P(base + 5);
        const float* g1 = P(base + 6); const float* g2 = P(base + 7);

        g = GemmArgs{};
        g.A = W1; g.a_bs = 0; g.a_ms = CS_; g.a_ks = 1;
        g.Bp = pf[i]; g.b_bs = (long long)CS_ * NF_; g.b_ks = NF_; g.b_ns = 1;
        g.D = U; g.d_bs = (long long)C_ * NF_; g.d_ms = NF_; g.d_ns = 1;
        g.M = C_; g.N = NF_; g.K = CS_; g.alpha = 1.0f;
        launch_gemm(stream, g, B_);

        gather_h1_kernel<<<gblocks, 256, 0, stream>>>(U, nnidx, nnw, b1, g1, be1, h1);

        g = GemmArgs{};
        g.A = W2; g.a_bs = 0; g.a_ms = C_; g.a_ks = 1;
        g.Bp = h1; g.b_bs = (long long)C_ * NP_; g.b_ks = NP_; g.b_ns = 1; g.b_bf16 = 1;
        g.D = interp; g.d_bs = (long long)C_ * NP_; g.d_ms = NP_; g.d_ns = 1;
        g.bias = b2; g.bn_g = g2; g.bn_be = be2; g.vec_axis = 0; g.do_gelu = 1;
        g.resid = interp; g.r_bs = (long long)C_ * NP_; g.r_ms = NP_; g.r_ns = 1;
        g.M = C_; g.N = NP_; g.K = C_; g.alpha = 1.0f;
        launch_gemm(stream, g, B_);
    }

    // ---- 7) final classifier ----
    g = GemmArgs{};
    g.A = interp; g.a_bs = (long long)C_ * NP_; g.a_ms = 1; g.a_ks = NP_;
    g.Bp = cls_W1; g.b_bs = 0; g.b_ks = 1; g.b_ns = C_;
    g.D = Hc; g.d_bs = (long long)NP_ * 256; g.d_ms = 256; g.d_ns = 1; g.d_bf16 = 1;
    g.bias = cls_b1; g.bn_g = cls_g1; g.bn_be = cls_be1; g.vec_axis = 1; g.do_gelu = 1;
    g.M = NP_; g.N = 256; g.K = C_; g.alpha = 1.0f;
    launch_gemm(stream, g, B_);

    g = GemmArgs{};
    g.A = Hc; g.a_bs = (long long)NP_ * 256; g.a_ms = 256; g.a_ks = 1; g.a_bf16 = 1;
    g.Bp = cls_W2; g.b_bs = 0; g.b_ks = 1; g.b_ns = 256;
    g.D = out_preds; g.d_bs = (long long)NP_ * NCLS_; g.d_ms = NCLS_; g.d_ns = 1;
    g.bias = cls_b2; g.vec_axis = 1;
    g.M = NP_; g.N = NCLS_; g.K = 256; g.alpha = 1.0f;
    launch_gemm(stream, g, B_);

    long long nrows = (long long)B_ * NP_;
    logsoftmax_kernel<<<(int)((nrows + 255) / 256), 256, 0, stream>>>(out_preds, out_logsm, nrows);
}
